// SwitchFeedForward_43035572305972
// MI455X (gfx1250) — matrix operand order repaired and checked by reading, not compiled, not run on hardware
//
#include <hip/hip_runtime.h>
#include <hip/hip_bf16.h>

// ---------------------------------------------------------------------------
// Switch (top-1 MoE) FFN for gfx1250 / MI455X.
// bf16 WMMA (v_wmma_f32_16x16x32_bf16) with f32 accumulation; gather-by-expert
// so each token runs exactly one expert's FFN (8x less work than reference).
// Weights are one-shot transpose-converted to bf16 [N][K] so both A and B
// tiles stage into LDS with b128 loads/stores; GEMM main loop is
// double-buffered (1 barrier / K-step) with global loads overlapping WMMA.
// All loop-invariant gather state (token row, base pointers) hoisted out of
// the K loop so the steady state is pure load/wmma/store.
// ---------------------------------------------------------------------------

typedef __bf16 bf16;
typedef __attribute__((ext_vector_type(4)))  __bf16 v4bf;
typedef __attribute__((ext_vector_type(8)))  __bf16 v8bf;
typedef __attribute__((ext_vector_type(16))) __bf16 v16bf;
typedef __attribute__((ext_vector_type(8)))  float  v8f;

#define TOKENS 8192
#define DIN    1024
#define DFF    4096
#define DOUT   1024
#define NEXP   8
#define BM     128
#define BN     128
#define BK     32
#define ASTR   40                      // LDS row stride in bf16 (80B: 16B aligned, conflict-free)
#define MAXBLK (TOKENS / BM + NEXP)    // 72 worst-case M-blocks across experts
#define RWG    (TOKENS / 256)          // 32 router workgroups

union Frag16 { v16bf v; v8bf h[2]; };

__device__ inline v8f wmma_bf16(v16bf a, v16bf b, v8f c) {
  // 8 args: (neg_a, A, neg_b, B, c_mod, C, reuse_a, reuse_b)
  return __builtin_amdgcn_wmma_f32_16x16x32_bf16(false, a, false, b, (short)0, c,
                                                 false, false);
}

// ---------------------------------------------------------------------------
__global__ void init_kernel(int* cursor) {
  if (threadIdx.x < NEXP) cursor[threadIdx.x] = 0;
}

// ---------------------------------------------------------------------------
// Router: logits = x @ switch_w + b, softmax, argmax; also x -> bf16 and
// per-workgroup count/prob partial sums (deterministic reduction later).
__global__ __launch_bounds__(256)
void router_kernel(const float* __restrict__ x, const float* __restrict__ sw,
                   const float* __restrict__ sb, int* __restrict__ routes,
                   float* __restrict__ pmax, bf16* __restrict__ xb,
                   int* __restrict__ wgCnt, float* __restrict__ wgProb) {
  __shared__ float lsw[DIN * NEXP];      // 32 KB: full switch_w
  __shared__ float allprob[256][NEXP];   // per-thread probs for ordered reduce
  __shared__ int   lcnt[NEXP];
  int tid = threadIdx.x;
  for (int i = tid; i < DIN * NEXP / 4; i += 256)
    ((float4*)lsw)[i] = ((const float4*)sw)[i];
  if (tid < NEXP) lcnt[tid] = 0;
  __syncthreads();

  long t = (long)blockIdx.x * 256 + tid;
  float acc[NEXP];
#pragma unroll
  for (int e = 0; e < NEXP; ++e) acc[e] = sb[e];
  const float* xr = x + t * DIN;
  bf16* xbr = xb + t * DIN;
  for (int d = 0; d < DIN; d += 4) {
    float4 xv = *(const float4*)(xr + d);
    v4bf xc;
    xc[0] = (bf16)xv.x; xc[1] = (bf16)xv.y; xc[2] = (bf16)xv.z; xc[3] = (bf16)xv.w;
    *(v4bf*)(xbr + d) = xc;
    const float* r0 = lsw + d * NEXP;
#pragma unroll
    for (int e = 0; e < NEXP; ++e)
      acc[e] += xv.x * r0[e] + xv.y * r0[NEXP + e] +
                xv.z * r0[2 * NEXP + e] + xv.w * r0[3 * NEXP + e];
  }
  float m = acc[0]; int idx = 0;
#pragma unroll
  for (int e = 1; e < NEXP; ++e)
    if (acc[e] > m) { m = acc[e]; idx = e; }   // first-max, matches argmax
  float pr[NEXP]; float sum = 0.f;
#pragma unroll
  for (int e = 0; e < NEXP; ++e) { pr[e] = __expf(acc[e] - m); sum += pr[e]; }
  float inv = 1.0f / sum;                      // = max softmax prob (exp(0)/sum)
#pragma unroll
  for (int e = 0; e < NEXP; ++e) allprob[tid][e] = pr[e] * inv;
  atomicAdd(&lcnt[idx], 1);                    // int atomics: deterministic
  routes[t] = idx;
  pmax[t]   = inv;
  __syncthreads();
  if (tid < NEXP) {
    float s = 0.f;
    for (int w = 0; w < 256; ++w) s += allprob[w][tid];  // fixed order
    wgProb[blockIdx.x * NEXP + tid] = s;
    wgCnt [blockIdx.x * NEXP + tid] = lcnt[tid];
  }
}

// ---------------------------------------------------------------------------
// One-shot transpose-convert: f32 src [E][R][C] -> bf16 dst [E][C][R].
// 32x32 tiles via LDS; coalesced on both sides.
__global__ __launch_bounds__(256)
void convt_kernel(const float* __restrict__ src, bf16* __restrict__ dst,
                  int R, int C) {
  __shared__ bf16 tile[32][33];
  long base = (long)blockIdx.z * R * C;
  int r0 = blockIdx.y * 32;   // K
  int c0 = blockIdx.x * 32;   // N
  int tr = threadIdx.x >> 5;  // 0..7
  int tc = threadIdx.x & 31;
#pragma unroll
  for (int i = 0; i < 4; ++i) {
    int r = tr + i * 8;
    tile[r][tc] = (bf16)src[base + (long)(r0 + r) * C + c0 + tc];
  }
  __syncthreads();
#pragma unroll
  for (int i = 0; i < 4; ++i) {
    int r = tr + i * 8;       // N index in dst
    dst[base + (long)(c0 + r) * R + r0 + tc] = tile[tc][r];
  }
}

// ---------------------------------------------------------------------------
// Single-thread scan: expert offsets, per-expert M-block starts, aux loss.
// meta[0..8] = token offsets, meta[16..24] = block starts.
__global__ void scan_kernel(const int* __restrict__ wgCnt,
                            const float* __restrict__ wgProb,
                            int* __restrict__ meta, float* __restrict__ aux_out) {
  int counts[NEXP]; float ps[NEXP];
  for (int e = 0; e < NEXP; ++e) {
    int c = 0; float p = 0.f;
    for (int w = 0; w < RWG; ++w) { c += wgCnt[w * NEXP + e]; p += wgProb[w * NEXP + e]; }
    counts[e] = c; ps[e] = p;
  }
  int off = 0, blk = 0;
  for (int e = 0; e < NEXP; ++e) {
    meta[e] = off; meta[16 + e] = blk;
    off += counts[e]; blk += (counts[e] + BM - 1) / BM;
  }
  meta[NEXP] = off; meta[16 + NEXP] = blk;
  float aux = 0.f;
  for (int e = 0; e < NEXP; ++e)
    aux += ((float)counts[e] / (float)TOKENS) * ps[e];
  aux_out[0] = (float)NEXP * aux;
}

// ---------------------------------------------------------------------------
__global__ void scatter_kernel(const int* __restrict__ routes,
                               const int* __restrict__ meta,
                               int* __restrict__ cursor, int* __restrict__ order) {
  int t = blockIdx.x * 256 + threadIdx.x;
  int e = routes[t];
  int pos = atomicAdd(&cursor[e], 1);
  order[meta[e] + pos] = t;   // permutation within expert; results invariant
}

// ---------------------------------------------------------------------------
// GEMM1: h[pos, DFF] = relu(Xe @ W1[e] + b1[e]), Xe gathered via order[].
// 128x128x32 tiles, 8 waves -> 2x4 wmma tiles each; double-buffered LDS.
__global__ __launch_bounds__(256)
void gemm1_kernel(const bf16* __restrict__ xb, const bf16* __restrict__ w1t,
                  const float* __restrict__ b1, const int* __restrict__ meta,
                  const int* __restrict__ order, bf16* __restrict__ hb) {
  __shared__ __align__(16) bf16 At[2][BM * ASTR];
  __shared__ __align__(16) bf16 Bt[2][BN * ASTR];
  __shared__ int rows[BM];
  int by = blockIdx.y;
  if (by >= meta[16 + NEXP]) return;
  int e = 0;
  while (e < NEXP - 1 && meta[17 + e] <= by) ++e;
  int mblk    = by - meta[16 + e];
  int posBase = meta[e] + mblk * BM;
  int posEnd  = meta[e + 1];
  int tid = threadIdx.x;
  if (tid < BM) {
    int p = posBase + tid;
    rows[tid] = (p < posEnd) ? order[p] : -1;
  }

  int lane = tid & 31, wave = tid >> 5;
  int waveM = wave & 3, waveN = wave >> 2;
  int nBase = blockIdx.x * BN;
  int lrow = lane & 15, lhalf = lane >> 4;
  int st_r = tid >> 1, st_p = tid & 1;       // staging: 2 threads per tile row
  const bf16* brow = w1t + ((long)e * DFF + nBase + st_r) * DIN + st_p * 16;
  v8f acc[2][4] = {};
  v8bf ra0 = {}, ra1 = {}, rb0, rb1;         // staged registers (zeros persist
                                             // for padded rows; loop-invariant)
  __syncthreads();                           // rows[] visible
  int tok = rows[st_r];                      // hoisted: invariant across K loop
  bool aval = tok >= 0;
  const bf16* arow = xb + (long)(aval ? tok : 0) * DIN + st_p * 16;

  auto loadG = [&](int kb) {
    if (aval) {
      const bf16* s = arow + kb;
      ra0 = *(const v8bf*)s; ra1 = *(const v8bf*)(s + 8);
    }
    const bf16* bs = brow + kb;
    rb0 = *(const v8bf*)bs; rb1 = *(const v8bf*)(bs + 8);
    __builtin_prefetch(bs + 8 * BK, 0, 1);   // 512B ahead along K stream
  };
  auto storeT = [&](int buf) {
    bf16* da = At[buf] + st_r * ASTR + st_p * 16;
    *(v8bf*)da = ra0; *(v8bf*)(da + 8) = ra1;
    bf16* db = Bt[buf] + st_r * ASTR + st_p * 16;
    *(v8bf*)db = rb0; *(v8bf*)(db + 8) = rb1;
  };

  loadG(0);
  storeT(0);
  __syncthreads();

  const int NK = DIN / BK;
  for (int i = 0; i < NK; ++i) {
    int cur = i & 1;
    if (i + 1 < NK) loadG((i + 1) * BK);     // overlap with WMMA below
    Frag16 a[2], b[4];
#pragma unroll
    for (int mi = 0; mi < 2; ++mi) {         // A frag: lane<16 K0..15; K16..31 else
      const bf16* ab = At[cur] + (waveM * 32 + mi * 16 + lrow) * ASTR;
      a[mi].h[0] = *(const v8bf*)(ab + lhalf * 16);
      a[mi].h[1] = *(const v8bf*)(ab + 8 + lhalf * 16);
    }
#pragma unroll
    for (int ni = 0; ni < 4; ++ni) {         // B frag: lane<16 K0..15; K16..31 else
      const bf16* bb = Bt[cur] + (waveN * 64 + ni * 16 + lrow) * ASTR + lhalf * 16;
      b[ni].h[0] = *(const v8bf*)bb;
      b[ni].h[1] = *(const v8bf*)(bb + 8);
    }
#pragma unroll
    for (int mi = 0; mi < 2; ++mi)
#pragma unroll
      for (int ni = 0; ni < 4; ++ni)
        acc[mi][ni] = wmma_bf16(a[mi].v, b[ni].v, acc[mi][ni]);
    if (i + 1 < NK) storeT(cur ^ 1);
    __syncthreads();
  }
  // epilogue: relu(acc + bias) -> bf16 h at sorted position
#pragma unroll
  for (int mi = 0; mi < 2; ++mi) {
    int pRow = posBase + waveM * 32 + mi * 16 + lhalf * 8;  // C: M = r + lhalf*8
#pragma unroll
    for (int ni = 0; ni < 4; ++ni) {
      int nCol = nBase + waveN * 64 + ni * 16 + lrow;       // C: N = lane&15
      float bias = b1[e * DFF + nCol];
#pragma unroll
      for (int r = 0; r < 8; ++r) {
        int p = pRow + r;
        if (p < posEnd) {
          float v = acc[mi][ni][r] + bias;
          hb[(long)p * DFF + nCol] = (bf16)(v > 0.f ? v : 0.f);
        }
      }
    }
  }
}

// ---------------------------------------------------------------------------
// GEMM2: y[tok] = (h @ W2[e] + b2[e]) * pmax[tok], scattered back to tokens.
__global__ __launch_bounds__(256)
void gemm2_kernel(const bf16* __restrict__ hb, const bf16* __restrict__ w2t,
                  const float* __restrict__ b2, const int* __restrict__ meta,
                  const int* __restrict__ order, const float* __restrict__ pmax,
                  float* __restrict__ y) {
  __shared__ __align__(16) bf16 At[2][BM * ASTR];
  __shared__ __align__(16) bf16 Bt[2][BN * ASTR];
  int by = blockIdx.y;
  if (by >= meta[16 + NEXP]) return;
  int e = 0;
  while (e < NEXP - 1 && meta[17 + e] <= by) ++e;
  int mblk    = by - meta[16 + e];
  int posBase = meta[e] + mblk * BM;
  int posEnd  = meta[e + 1];
  int tid = threadIdx.x;
  int lane = tid & 31, wave = tid >> 5;
  int waveM = wave & 3, waveN = wave >> 2;
  int nBase = blockIdx.x * BN;
  int lrow = lane & 15, lhalf = lane >> 4;
  int st_r = tid >> 1, st_p = tid & 1;
  int ap = posBase + st_r;
  bool aval = ap < posEnd;
  const bf16* arow = hb + (long)(aval ? ap : 0) * DFF + st_p * 16;
  const bf16* brow = w2t + ((long)e * DOUT + nBase + st_r) * DFF + st_p * 16;
  v8f acc[2][4] = {};
  v8bf ra0 = {}, ra1 = {}, rb0, rb1;

  auto loadG = [&](int kb) {
    if (aval) {
      const bf16* s = arow + kb;
      ra0 = *(const v8bf*)s; ra1 = *(const v8bf*)(s + 8);
    }
    const bf16* bs = brow + kb;
    rb0 = *(const v8bf*)bs; rb1 = *(const v8bf*)(bs + 8);
    __builtin_prefetch(bs + 8 * BK, 0, 1);
  };
  auto storeT = [&](int buf) {
    bf16* da = At[buf] + st_r * ASTR + st_p * 16;
    *(v8bf*)da = ra0; *(v8bf*)(da + 8) = ra1;
    bf16* db = Bt[buf] + st_r * ASTR + st_p * 16;
    *(v8bf*)db = rb0; *(v8bf*)(db + 8) = rb1;
  };

  loadG(0);
  storeT(0);
  __syncthreads();

  const int NK = DFF / BK;
  for (int i = 0; i < NK; ++i) {
    int cur = i & 1;
    if (i + 1 < NK) loadG((i + 1) * BK);
    Frag16 a[2], b[4];
#pragma unroll
    for (int mi = 0; mi < 2; ++mi) {
      const bf16* ab = At[cur] + (waveM * 32 + mi * 16 + lrow) * ASTR;
      a[mi].h[0] = *(const v8bf*)(ab + lhalf * 16);
      a[mi].h[1] = *(const v8bf*)(ab + 8 + lhalf * 16);
    }
#pragma unroll
    for (int ni = 0; ni < 4; ++ni) {
      const bf16* bb = Bt[cur] + (waveN * 64 + ni * 16 + lrow) * ASTR + lhalf * 16;
      b[ni].h[0] = *(const v8bf*)bb;
      b[ni].h[1] = *(const v8bf*)(bb + 8);
    }
#pragma unroll
    for (int mi = 0; mi < 2; ++mi)
#pragma unroll
      for (int ni = 0; ni < 4; ++ni)
        acc[mi][ni] = wmma_bf16(a[mi].v, b[ni].v, acc[mi][ni]);
    if (i + 1 < NK) storeT(cur ^ 1);
    __syncthreads();
  }
  // epilogue: bias + pmax scale, scatter to original token rows (f32, streaming)
#pragma unroll
  for (int mi = 0; mi < 2; ++mi) {
    int pRow = posBase + waveM * 32 + mi * 16 + lhalf * 8;
#pragma unroll
    for (int r = 0; r < 8; ++r) {
      int p = pRow + r;
      if (p >= posEnd) continue;
      int tok = order[p];
      float sc = pmax[tok];
      float* yr = y + (long)tok * DOUT;
#pragma unroll
      for (int ni = 0; ni < 4; ++ni) {
        int nCol = nBase + waveN * 64 + ni * 16 + lrow;
        __builtin_nontemporal_store((acc[mi][ni][r] + b2[e * DOUT + nCol]) * sc,
                                    yr + nCol);
      }
    }
  }
}

// ---------------------------------------------------------------------------
extern "C" void kernel_launch(void* const* d_in, const int* in_sizes, int n_in,
                              void* d_out, int out_size, void* d_ws, size_t ws_size,
                              hipStream_t stream) {
  (void)in_sizes; (void)n_in; (void)out_size; (void)ws_size;
  const float* x  = (const float*)d_in[0];
  const float* sw = (const float*)d_in[1];
  const float* sb = (const float*)d_in[2];
  const float* w1 = (const float*)d_in[3];
  const float* b1 = (const float*)d_in[4];
  const float* w2 = (const float*)d_in[5];
  const float* b2 = (const float*)d_in[6];
  float* out = (float*)d_out;

  // Workspace layout (256B aligned). Total ~210 MB.
  char* ws = (char*)d_ws;
  size_t off = 0;
  auto alloc = [&](size_t bytes) {
    size_t r = off; off += (bytes + 255) & ~(size_t)255; return r;
  };
  int*   routes = (int*)  (ws + alloc((size_t)TOKENS * 4));
  float* pmax   = (float*)(ws + alloc((size_t)TOKENS * 4));
  int*   order  = (int*)  (ws + alloc((size_t)TOKENS * 4));
  int*   meta   = (int*)  (ws + alloc(64 * 4));
  int*   cursor = (int*)  (ws + alloc(NEXP * 4));
  int*   wgCnt  = (int*)  (ws + alloc((size_t)RWG * NEXP * 4));
  float* wgProb = (float*)(ws + alloc((size_t)RWG * NEXP * 4));
  bf16*  xb     = (bf16*) (ws + alloc((size_t)TOKENS * DIN * 2));
  bf16*  hb     = (bf16*) (ws + alloc((size_t)TOKENS * DFF * 2));
  bf16*  w1t    = (bf16*) (ws + alloc((size_t)NEXP * DIN * DFF * 2));
  bf16*  w2t    = (bf16*) (ws + alloc((size_t)NEXP * DFF * DOUT * 2));

  init_kernel<<<1, 32, 0, stream>>>(cursor);
  router_kernel<<<RWG, 256, 0, stream>>>(x, sw, sb, routes, pmax, xb, wgCnt, wgProb);
  // w1 [E][DIN][DFF] -> w1t [E][DFF][DIN]; w2 [E][DFF][DOUT] -> w2t [E][DOUT][DFF]
  convt_kernel<<<dim3(DFF / 32, DIN / 32, NEXP), 256, 0, stream>>>(w1, w1t, DIN, DFF);
  convt_kernel<<<dim3(DOUT / 32, DFF / 32, NEXP), 256, 0, stream>>>(w2, w2t, DFF, DOUT);
  scan_kernel<<<1, 1, 0, stream>>>(wgCnt, wgProb, meta, out + (size_t)TOKENS * DOUT);
  scatter_kernel<<<TOKENS / 256, 256, 0, stream>>>(routes, meta, cursor, order);
  gemm1_kernel<<<dim3(DFF / BN, MAXBLK), 256, 0, stream>>>(xb, w1t, b1, meta, order, hb);
  gemm2_kernel<<<dim3(DOUT / BN, MAXBLK), 256, 0, stream>>>(hb, w2t, b2, meta, order,
                                                            pmax, out);
}
